// AttentionGrowingModule_76570676953695
// MI455X (gfx1250) — matrix operand order repaired and checked by reading, not compiled, not run on hardware
//
#include <hip/hip_runtime.h>

typedef __attribute__((ext_vector_type(16))) _Float16 v16h;
typedef __attribute__((ext_vector_type(8)))  _Float16 v8h;
typedef __attribute__((ext_vector_type(4)))  _Float16 v4h;
typedef __attribute__((ext_vector_type(8)))  float    v8f;

union AFrag { v16h v; v8h h[2]; };

#define D_E 1024
#define D_K 128
#define SEQ 2048
#define NBATCH 8
#define NTOK (NBATCH * SEQ)          // 16384
// log2(e) / sqrt(128) folded into Q so softmax uses exp2
#define QK_SCALE 0.12751741f

#if defined(__has_builtin)
#if __has_builtin(__builtin_amdgcn_sched_barrier)
#define SCHED_BARRIER() __builtin_amdgcn_sched_barrier(0)
#endif
#endif
#ifndef SCHED_BARRIER
#define SCHED_BARRIER() asm volatile("" ::: "memory")
#endif

// wait for all outstanding async (ASYNCcnt) LDS-DMA transfers
#define ASYNC_WAIT() asm volatile("s_wait_asynccnt 0" ::: "memory")

typedef __attribute__((address_space(3))) void lds_void;

// raw 32-bit LDS byte offset of a generic pointer into shared memory
__device__ __forceinline__ unsigned lds_addr(const void* p) {
  return (unsigned)(size_t)(lds_void*)p;
}

// one-lane 16-byte async DMA: global -> LDS, tracked by ASYNCcnt
__device__ __forceinline__ void async_b128(_Float16* lds_dst,
                                           const _Float16* gsrc) {
  unsigned l = lds_addr(lds_dst);
  asm volatile("global_load_async_to_lds_b128 %0, %1, off"
               :: "v"(l), "v"(gsrc) : "memory");
}

__device__ __forceinline__ v8f wmma16(v16h a, v16h b, v8f c) {
  return __builtin_amdgcn_wmma_f32_16x16x32_f16(false, a, false, b, (short)0, c,
                                                false, false);
}

// A-fragment: 16x32 f16 tile from row-major matrix, row stride ld (halves).
__device__ __forceinline__ v16h load_a16x32(const _Float16* base, int ld, int lane) {
  int m  = lane & 15;
  int hi = lane >> 4;
  const _Float16* p = base + m * ld + hi * 8;
  AFrag f;
  f.h[0] = *(const v8h*)(p);
  f.h[1] = *(const v8h*)(p + 16);
  return f.v;
}

// B-fragment: 32x16 (k x n) read from n-major storage BT[n][k], stride ld.
__device__ __forceinline__ v16h load_b32x16(const _Float16* baseT, int ld, int lane) {
  int n  = lane & 15;
  int hi = lane >> 4;
  const _Float16* p = baseT + n * ld + hi * 8;
  AFrag f;
  f.h[0] = *(const v8h*)(p);
  f.h[1] = *(const v8h*)(p + 16);
  return f.v;
}

// ---------------- prep: fp32 -> f16 conversions -------------------------

__global__ __launch_bounds__(256) void cvt_kernel(const float* __restrict__ src,
                                                  _Float16* __restrict__ dst, int n) {
  int i = (blockIdx.x * 256 + threadIdx.x) * 4;
  if (i < n) {
    float4 v = *(const float4*)(src + i);
    v4h o = { (_Float16)v.x, (_Float16)v.y, (_Float16)v.z, (_Float16)v.w };
    *(v4h*)(dst + i) = o;
  }
}

__global__ __launch_bounds__(256) void transpose_cvt_kernel(const float* __restrict__ src,
                                                            _Float16* __restrict__ dst,
                                                            int R, int C) {
  int i = blockIdx.x * 256 + threadIdx.x;
  if (i < R * C) {
    int r = i / C, c = i % C;
    dst[(size_t)c * R + r] = (_Float16)src[i];
  }
}

// ---------------- QKV projection ---------------------------------------
// grid (256, 3), block 128 (4 waves). wave -> 16 token rows x 128 out cols.
// Weight k-chunks async-DMA'd into double-buffered LDS.
__global__ __launch_bounds__(128) void qkv_kernel(
    const _Float16* __restrict__ Xh,
    const _Float16* __restrict__ WQt, const _Float16* __restrict__ WKt,
    const _Float16* __restrict__ WVt,
    const float* __restrict__ bQ, const float* __restrict__ bK,
    const float* __restrict__ bV,
    _Float16* __restrict__ Qh, _Float16* __restrict__ Kh,
    _Float16* __restrict__ Vt) {
  __shared__ alignas(32) _Float16 Wl[2 * 128 * 40];   // 2 x (128 rows, pad 40)
  __shared__ alignas(32) _Float16 Sg[4 * 128 * 24];   // per-wave epilogue staging

  int lane = threadIdx.x & 31;
  int wid  = threadIdx.x >> 5;
  int job  = blockIdx.y;                     // 0=Q 1=K 2=V
  const _Float16* Wt  = (job == 0) ? WQt : (job == 1) ? WKt : WVt;
  const float*   bias = (job == 0) ? bQ  : (job == 1) ? bK  : bV;

  int mt = blockIdx.x * 4 + wid;             // 0..1023
  int m0 = mt * 16;
  const _Float16* Xrow = Xh + (size_t)m0 * D_E;

  // 128 rows x 32 halves = 512 units of 8 halves (16 B)
  auto asyncW = [&](int c, int buf) {
#pragma unroll
    for (int u = threadIdx.x; u < 512; u += 128) {
      int row = u >> 2, seg = u & 3;
      async_b128(Wl + buf * (128 * 40) + row * 40 + seg * 8,
                 Wt + (size_t)row * D_E + c * 32 + seg * 8);
    }
  };

  v8f acc[8] = {};
  asyncW(0, 0);
  v16h a_cur = load_a16x32(Xrow, D_E, lane);
  ASYNC_WAIT();
  __syncthreads();

  for (int c = 0; c < 32; ++c) {             // e-dim in chunks of 32
    int cur = c & 1, nxt = cur ^ 1;
    int cn = (c + 1 < 32) ? c + 1 : c;
    asyncW(cn, nxt);                         // DMA overlaps this chunk's math
    v16h a_nxt = load_a16x32(Xrow + cn * 32, D_E, lane);
    SCHED_BARRIER();
    const _Float16* Wc = Wl + cur * (128 * 40);
    v16h bf[8];
#pragma unroll
    for (int t = 0; t < 8; ++t) bf[t] = load_b32x16(Wc + t * 16 * 40, 40, lane);
#pragma unroll
    for (int t = 0; t < 8; ++t) acc[t] = wmma16(a_cur, bf[t], acc[t]);
    a_cur = a_nxt;
    ASYNC_WAIT();
    __syncthreads();
  }

  int n_lo = lane & 15, hi = lane >> 4;

  if (job < 2) {
    float scl = (job == 0) ? QK_SCALE : 1.0f;
    _Float16* S = Sg + wid * (128 * 24);     // use 16 rows, stride 136
#pragma unroll
    for (int t = 0; t < 8; ++t) {
      float bv = bias[t * 16 + n_lo];
#pragma unroll
      for (int r = 0; r < 8; ++r)
        S[(r + 8 * hi) * 136 + t * 16 + n_lo] =
            (_Float16)((acc[t][r] + bv) * scl);
    }
    _Float16* O = ((job == 0) ? Qh : Kh) + (size_t)m0 * D_K;
    int row = lane >> 1, cb = (lane & 1) * 64;
    const v8h* src = (const v8h*)(S + row * 136 + cb);
    v8h* dst = (v8h*)(O + (size_t)row * D_K + cb);
#pragma unroll
    for (int j = 0; j < 8; ++j) dst[j] = src[j];
  } else {
    // transpose in LDS: v-major 128 rows x 16, pad stride 24
    _Float16* S = Sg + wid * (128 * 24);
#pragma unroll
    for (int t = 0; t < 8; ++t) {
      float bv = bias[t * 16 + n_lo];
#pragma unroll
      for (int r = 0; r < 8; ++r)
        S[(t * 16 + n_lo) * 24 + (r + 8 * hi)] = (_Float16)(acc[t][r] + bv);
    }
    int b = m0 >> 11, s0 = m0 & 2047;
#pragma unroll
    for (int j = 0; j < 4; ++j) {
      int v = j * 32 + lane;
      v8h lo = *(const v8h*)(S + v * 24);
      v8h hi8 = *(const v8h*)(S + v * 24 + 8);
      _Float16* dp = Vt + ((size_t)b * D_K + v) * SEQ + s0;
      *(v8h*)(dp) = lo;
      *(v8h*)(dp + 8) = hi8;
    }
  }
}

// ---------------- flash attention --------------------------------------
// grid 256, block 128 (4 waves, same batch). wave -> 16 queries x d_v=128.
// K/V tiles async-DMA'd into double-buffered LDS; s_wait_asynccnt only at
// the bottom of the iteration, so the DMA overlaps the whole tile compute.
__global__ __launch_bounds__(128) void attn_kernel(
    const _Float16* __restrict__ Qh, const _Float16* __restrict__ Kh,
    const _Float16* __restrict__ Vt, _Float16* __restrict__ Hh) {
  __shared__ alignas(32) _Float16 Klds[2 * 32 * 136];   // 17408 B
  __shared__ alignas(32) _Float16 Vlds[2 * 128 * 40];   // 20480 B
  __shared__ alignas(32) _Float16 Plds[4 * 16 * 40];    // per-wave P staging

  int lane = threadIdx.x & 31;
  int wid  = threadIdx.x >> 5;
  int qt   = blockIdx.x * 4 + wid;           // 0..1023
  int b    = qt >> 7;                        // 128 q-tiles per batch
  int m0   = qt * 16;
  const _Float16* Qbase = Qh + (size_t)m0 * D_K;
  const _Float16* Kbase = Kh + (size_t)b * SEQ * D_K;
  const _Float16* Vbase = Vt + (size_t)b * D_K * SEQ;
  _Float16* Pw = Plds + wid * (16 * 40);

  // K tile: 32 rows x 128 halves = 512 units of 8 halves
  auto asyncK = [&](int kt, int buf) {
#pragma unroll
    for (int u = threadIdx.x; u < 512; u += 128) {
      int row = u >> 4, seg = u & 15;
      async_b128(Klds + buf * (32 * 136) + row * 136 + seg * 8,
                 Kbase + (size_t)(kt * 32 + row) * D_K + seg * 8);
    }
  };
  // V tile: 128 rows x 32 halves = 512 units of 8 halves
  auto asyncV = [&](int kt, int buf) {
#pragma unroll
    for (int u = threadIdx.x; u < 512; u += 128) {
      int row = u >> 2, seg = u & 3;
      async_b128(Vlds + buf * (128 * 40) + row * 40 + seg * 8,
                 Vbase + (size_t)row * SEQ + kt * 32 + seg * 8);
    }
  };

  // Q fragments held in registers (16 x 128 = 4 A-frags)
  v16h qf[4];
#pragma unroll
  for (int c = 0; c < 4; ++c)
    qf[c] = load_a16x32(Qbase + c * 32, D_K, lane);

  v8f acc[8] = {};
  float rmax[8], rsum[8];
#pragma unroll
  for (int r = 0; r < 8; ++r) { rmax[r] = -1e30f; rsum[r] = 0.0f; }

  int n_lo = lane & 15, hi = lane >> 4;

  asyncK(0, 0);
  asyncV(0, 0);
  ASYNC_WAIT();
  __syncthreads();

  for (int kt = 0; kt < SEQ / 32; ++kt) {
    int cur = kt & 1, nxt = cur ^ 1;
    int ktn = (kt + 1 < SEQ / 32) ? kt + 1 : kt;
    asyncK(ktn, nxt);                        // DMA next tile, no wait here
    asyncV(ktn, nxt);
    SCHED_BARRIER();

    // S(16x32) = Q . K^T from LDS (batched ds loads, then 8 WMMAs)
    const _Float16* Kc = Klds + cur * (32 * 136);
    v16h kf[8];
#pragma unroll
    for (int j = 0; j < 2; ++j)
#pragma unroll
      for (int c = 0; c < 4; ++c)
        kf[j * 4 + c] = load_b32x16(Kc + j * 16 * 136 + c * 32, 136, lane);
    v8f s0 = {}, s1 = {};
#pragma unroll
    for (int c = 0; c < 4; ++c) {
      s0 = wmma16(qf[c], kf[c], s0);
      s1 = wmma16(qf[c], kf[4 + c], s1);
    }

    // online softmax: row reductions across 16-lane halves
    float nm[8], al[8];
#pragma unroll
    for (int r = 0; r < 8; ++r) {
      float v = fmaxf(s0[r], s1[r]);
      v = fmaxf(v, __shfl_xor(v, 1, 16));
      v = fmaxf(v, __shfl_xor(v, 2, 16));
      v = fmaxf(v, __shfl_xor(v, 4, 16));
      v = fmaxf(v, __shfl_xor(v, 8, 16));
      nm[r] = fmaxf(rmax[r], v);
      al[r] = exp2f(rmax[r] - nm[r]);
      rmax[r] = nm[r];
    }

    // P = exp2(S - max); row sums; P -> LDS row-major (stride 40)
#pragma unroll
    for (int r = 0; r < 8; ++r) {
      float p0 = exp2f(s0[r] - nm[r]);
      float p1 = exp2f(s1[r] - nm[r]);
      float ts = p0 + p1;
      ts += __shfl_xor(ts, 1, 16);
      ts += __shfl_xor(ts, 2, 16);
      ts += __shfl_xor(ts, 4, 16);
      ts += __shfl_xor(ts, 8, 16);
      rsum[r] = rsum[r] * al[r] + ts;
      int m = r + 8 * hi;
      Pw[m * 40 + n_lo]      = (_Float16)p0;
      Pw[m * 40 + 16 + n_lo] = (_Float16)p1;
    }

    // rescale running accumulator
#pragma unroll
    for (int t = 0; t < 8; ++t)
#pragma unroll
      for (int r = 0; r < 8; ++r) acc[t][r] *= al[r];

    // re-read P as an A-fragment (wave-local LDS, no barrier needed)
    AFrag pa;
    {
      const _Float16* p = Pw + (lane & 15) * 40 + hi * 8;
      pa.h[0] = *(const v8h*)(p);
      pa.h[1] = *(const v8h*)(p + 16);
    }

    // H += P . V from LDS (batched ds loads, then 8 WMMAs)
    const _Float16* Vc = Vlds + cur * (128 * 40);
    v16h vf[8];
#pragma unroll
    for (int t = 0; t < 8; ++t)
      vf[t] = load_b32x16(Vc + t * 16 * 40, 40, lane);
#pragma unroll
    for (int t = 0; t < 8; ++t) acc[t] = wmma16(pa.v, vf[t], acc[t]);

    ASYNC_WAIT();      // next tile's DMA must land before the barrier
    __syncthreads();
  }

  // normalize, stage 16x128 f16 tile (reuse Klds, stride 136), copy out
  float inv[8];
#pragma unroll
  for (int r = 0; r < 8; ++r) inv[r] = 1.0f / rsum[r];
  _Float16* S = Klds + wid * (16 * 136);
#pragma unroll
  for (int t = 0; t < 8; ++t)
#pragma unroll
    for (int r = 0; r < 8; ++r)
      S[(r + 8 * hi) * 136 + t * 16 + n_lo] = (_Float16)(acc[t][r] * inv[r]);

  int row = lane >> 1, cb = (lane & 1) * 64;
  const v8h* src = (const v8h*)(S + row * 136 + cb);
  v8h* dst = (v8h*)(Hh + (size_t)(m0 + row) * D_K + cb);
#pragma unroll
  for (int j = 0; j < 8; ++j) dst[j] = src[j];
}

// ---------------- output projection + residual -------------------------
// grid 2048, block 128. 4 waves share one 16-row H tile (async DMA to LDS);
// each wave does a different 128-col group; weight frags double-buffered.
__global__ __launch_bounds__(128) void oproj_kernel(
    const _Float16* __restrict__ Hh, const _Float16* __restrict__ WOt,
    const float* __restrict__ bO, const float* __restrict__ X,
    float* __restrict__ Out) {
  __shared__ alignas(32) _Float16 Hl[16 * 136];       // shared A tile
  __shared__ alignas(16) float    SF[4 * 16 * 132];   // per-wave f32 staging

  int lane = threadIdx.x & 31;
  int wid  = threadIdx.x >> 5;
  int wg   = blockIdx.x * 4 + wid;           // 0..8191
  int mt   = wg >> 3;
  int g    = wg & 7;
  int m0 = mt * 16, e0 = g * 128;

  // async DMA of the H tile: 16 rows x 128 halves = 256 units of 8 halves
#pragma unroll
  for (int u = threadIdx.x; u < 256; u += 128) {
    int row = u >> 4, seg = u & 15;
    async_b128(Hl + row * 136 + seg * 8,
               Hh + (size_t)(m0 + row) * D_K + seg * 8);
  }

  v8f acc[8] = {};
  v16h bcur[8];
#pragma unroll
  for (int t = 0; t < 8; ++t)
    bcur[t] = load_b32x16(WOt + (size_t)(e0 + t * 16) * D_K, D_K, lane);

  ASYNC_WAIT();
  __syncthreads();

#pragma unroll
  for (int c = 0; c < 4; ++c) {
    int cn = (c + 1 < 4) ? c + 1 : c;
    v16h bnxt[8];
#pragma unroll
    for (int t = 0; t < 8; ++t)
      bnxt[t] = load_b32x16(WOt + (size_t)(e0 + t * 16) * D_K + cn * 32, D_K, lane);
    SCHED_BARRIER();
    v16h a = load_a16x32(Hl + c * 32, 136, lane);
#pragma unroll
    for (int t = 0; t < 8; ++t) acc[t] = wmma16(a, bcur[t], acc[t]);
#pragma unroll
    for (int t = 0; t < 8; ++t) bcur[t] = bnxt[t];
  }

  int n_lo = lane & 15, hi = lane >> 4;
  float* Sw = SF + wid * (16 * 132);
#pragma unroll
  for (int t = 0; t < 8; ++t) {
    float bv = bO[e0 + t * 16 + n_lo];
#pragma unroll
    for (int r = 0; r < 8; ++r)
      Sw[(r + 8 * hi) * 132 + t * 16 + n_lo] = acc[t][r] + bv;
  }

  // coalesced residual add + store: 2 lanes per row, 64 floats each
  int row = lane >> 1, cb = (lane & 1) * 64;
  const float4* src = (const float4*)(Sw + row * 132 + cb);
  size_t base = (size_t)(m0 + row) * D_E + e0 + cb;
#pragma unroll
  for (int j = 0; j < 16; ++j) {
    float4 x = *(const float4*)(X + base + j * 4);
    float4 s = src[j];
    float4 o;
    o.x = x.x + s.x; o.y = x.y + s.y; o.z = x.z + s.z; o.w = x.w + s.w;
    *(float4*)(Out + base + j * 4) = o;
  }
}

// ---------------- launch -----------------------------------------------

extern "C" void kernel_launch(void* const* d_in, const int* in_sizes, int n_in,
                              void* d_out, int out_size, void* d_ws, size_t ws_size,
                              hipStream_t stream) {
  (void)in_sizes; (void)n_in; (void)out_size; (void)ws_size;
  const float* X  = (const float*)d_in[0];
  const float* WQ = (const float*)d_in[1];
  const float* bQ = (const float*)d_in[2];
  const float* WK = (const float*)d_in[3];
  const float* bK = (const float*)d_in[4];
  const float* WV = (const float*)d_in[5];
  const float* bV = (const float*)d_in[6];
  const float* WO = (const float*)d_in[7];
  const float* bO = (const float*)d_in[8];
  float* Out = (float*)d_out;

  char* ws = (char*)d_ws;
  size_t off = 0;
  auto take = [&](size_t bytes) { char* p = ws + off; off += bytes; return p; };
  _Float16* Xh  = (_Float16*)take((size_t)NTOK * D_E * 2);   // 32 MB
  _Float16* Qh  = (_Float16*)take((size_t)NTOK * D_K * 2);   // 4 MB
  _Float16* Kh  = (_Float16*)take((size_t)NTOK * D_K * 2);
  _Float16* Vt  = (_Float16*)take((size_t)NTOK * D_K * 2);   // transposed per batch
  _Float16* Hh  = (_Float16*)take((size_t)NTOK * D_K * 2);
  _Float16* WQt = (_Float16*)take((size_t)D_E * D_K * 2);
  _Float16* WKt = (_Float16*)take((size_t)D_E * D_K * 2);
  _Float16* WVt = (_Float16*)take((size_t)D_E * D_K * 2);
  _Float16* WOt = (_Float16*)take((size_t)D_E * D_K * 2);

  cvt_kernel<<<NTOK * D_E / 1024, 256, 0, stream>>>(X, Xh, NTOK * D_E);
  transpose_cvt_kernel<<<512, 256, 0, stream>>>(WQ, WQt, D_E, D_K);
  transpose_cvt_kernel<<<512, 256, 0, stream>>>(WK, WKt, D_E, D_K);
  transpose_cvt_kernel<<<512, 256, 0, stream>>>(WV, WVt, D_E, D_K);
  transpose_cvt_kernel<<<512, 256, 0, stream>>>(WO, WOt, D_K, D_E);

  qkv_kernel<<<dim3(256, 3), 128, 0, stream>>>(Xh, WQt, WKt, WVt, bQ, bK, bV,
                                               Qh, Kh, Vt);
  attn_kernel<<<256, 128, 0, stream>>>(Qh, Kh, Vt, Hh);
  oproj_kernel<<<2048, 128, 0, stream>>>(Hh, WOt, bO, X, Out);
}
